// ALSHConv2d_78855599555126
// MI455X (gfx1250) — matrix operand order repaired and checked by reading, not compiled, not run on hardware
//
#include <hip/hip_runtime.h>

#define N_      16
#define C_      256
#define H_      56
#define W_      56
#define O_      512
#define KS_     3
#define SPAN_   (KS_*KS_*C_)      // 2304
#define TABLE_  16
#define BUCKET_ 32
#define M_AUG_  5
#define HW_     (H_*W_)           // 3136
#define PIX_PER_WG 128
#define WGS_PER_IMG 25            // ceil(3136/128)
#define HALO_ROWS 6
#define HALO_W 58
#define HALO_ELEMS (HALO_ROWS*HALO_W*32)  // 11136 bf16 (21.75 KB)
#define WLINES  (BUCKET_*KS_*KS_)         // 288 lines of 32 bf16
#define WSLAB   (WLINES*32)               // 9216 bf16 = 18 KB per buffer

typedef __attribute__((ext_vector_type(16))) __bf16 v16bf;
typedef __attribute__((ext_vector_type(8)))  float  v8f;
typedef __attribute__((ext_vector_type(4)))  unsigned int u32x4;
typedef __attribute__((ext_vector_type(8)))  int  i32x8;
typedef __attribute__((ext_vector_type(4)))  int  i32x4;

struct U32x8 { uint4 lo; uint4 hi; };
union  ABfrag { U32x8 u; v16bf v; };

#if __has_builtin(__builtin_amdgcn_s_wait_tensorcnt)
#define WAIT_TENSORCNT(n) __builtin_amdgcn_s_wait_tensorcnt(n)
#else
#define WAIT_TENSORCNT(n) asm volatile("s_wait_tensorcnt %0" :: "n"(n) : "memory")
#endif

__device__ __forceinline__ unsigned short f2bf(float f) {
    unsigned int u = __float_as_uint(f);
    unsigned int r = u + 0x7FFFu + ((u >> 16) & 1u);   // round-to-nearest-even
    return (unsigned short)(r >> 16);
}

__device__ __forceinline__ unsigned lds_byte_offset(const void* p) {
    return (unsigned)(size_t)(__attribute__((address_space(3))) const void*)p;
}

// TDM: 2D tile, 288 lines x 32 bf16, line stride 256 elements, into LDS.
// D# per CDNA5 ISA: group0 {count=1, lds_addr,
// global_addr[56:0], type=2}; group1 {data_size=2B, tensor_dim0=32,
// tensor_dim1=288, tile_dim0=32, tile_dim1=288, dim0_stride=256}.
__device__ __forceinline__ void tdm_load_wslab(unsigned lds_off,
                                               const unsigned short* gsrc) {
    const unsigned long long ga = (unsigned long long)(size_t)gsrc;
    u32x4 g0;
    g0[0] = 1u;                                        // count=1, no gather
    g0[1] = lds_off;                                   // LDS byte address
    g0[2] = (unsigned)ga;                              // global addr [31:0]
    g0[3] = (unsigned)((ga >> 32) & 0x01FFFFFFu) | (2u << 30); // [56:32]|type=2
    i32x8 g1;
    g1[0] = (int)(1u << 16);                           // data_size = 2 bytes
    g1[1] = (int)(32u << 16);                          // tensor_dim0 = 32
    g1[2] = (int)(288u << 16);                         // tensor_dim1 = 288
    g1[3] = (int)(32u << 16);                          // tile_dim0 = 32
    g1[4] = 288;                                       // tile_dim1=288, tile_dim2=0
    g1[5] = 256;                                       // tensor_dim0_stride = 256
    g1[6] = 0;
    g1[7] = 0;
#if __has_builtin(__builtin_amdgcn_tensor_load_to_lds)
    const i32x4 z4 = {0, 0, 0, 0};                     // groups 2/3 unused (2D)
    const i32x8 z8 = {0, 0, 0, 0, 0, 0, 0, 0};
    __builtin_amdgcn_tensor_load_to_lds(g0, g1, z4, z4, z8, 0);
#else
    asm volatile("tensor_load_to_lds %0, %1" :: "s"(g0), "s"(g1) : "memory");
#endif
}

// ---------------------------------------------------------------------------
// Kernel 1: hash -> bucket row list -> gather selected kernel rows as bf16.
// ---------------------------------------------------------------------------
__global__ __launch_bounds__(256)
void alsh_select_pack(const float* __restrict__ x,
                      const float* __restrict__ kernels,
                      const float* __restrict__ a,
                      const int*   __restrict__ table,
                      int*          rows_ws,
                      unsigned short* __restrict__ wpack)
{
    __shared__ float s_dot[256];
    __shared__ float s_nrm[256];
    __shared__ int   s_rows[BUCKET_];

    const int tid = threadIdx.x;

    float dot = 0.f, nrm = 0.f;
    for (int s = tid; s < SPAN_; s += 256) {
        const int k  = s / C_;
        const int c  = s - k * C_;
        const int iy = k / KS_ - 1;
        const int ix = k % KS_ - 1;
        float v = 0.f;
        if (iy >= 0 && ix >= 0)
            v = x[(c * H_ + iy) * W_ + ix];     // image n = 0, output pixel (0,0)
        dot += a[s] * v;
        nrm += v * v;
    }
    s_dot[tid] = dot; s_nrm[tid] = nrm;
    __syncthreads();
    for (int off = 128; off > 0; off >>= 1) {
        if (tid < off) { s_dot[tid] += s_dot[tid + off]; s_nrm[tid] += s_nrm[tid + off]; }
        __syncthreads();
    }
    if (tid == 0) {
        float asum = 0.f;
        for (int m = 0; m < M_AUG_; ++m) asum += a[SPAN_ + m];
        const float val = s_dot[0] / sqrtf(s_nrm[0]) + 0.5f * asum;
        const int idx = ((int)fabsf(floorf(val))) % TABLE_;
        for (int b = 0; b < BUCKET_; ++b) {
            const int r = table[idx * BUCKET_ + b];
            s_rows[b]  = r;
            rows_ws[b] = r;
        }
    }
    __syncthreads();

    for (int e = tid; e < BUCKET_ * SPAN_; e += 256) {
        const int r  = e / SPAN_;
        const int kk = e - r * SPAN_;
        wpack[e] = f2bf(kernels[s_rows[r] * SPAN_ + kk]);
    }
}

// ---------------------------------------------------------------------------
// Kernel 2: implicit-GEMM conv, halo-tiled, TDM-fed weights, bf16 WMMA.
// Per channel-block: TDM DMA of next weight slab overlaps vector-pipe halo
// staging and the 9-tap x 2-pixel-tile WMMA chain of the current block.
// ---------------------------------------------------------------------------
__global__ __launch_bounds__(256)
void alsh_conv_wmma(const float* __restrict__ x,
                    const unsigned short* __restrict__ wpack,
                    float* __restrict__ out)
{
    __shared__ __align__(16) unsigned short s_t[HALO_ELEMS];
    __shared__ __align__(16) unsigned short s_w[2][WSLAB];   // double-buffer

    const int tid    = threadIdx.x;
    const int blk    = blockIdx.x;
    const int n      = blk / WGS_PER_IMG;
    const int ptbase = (blk - n * WGS_PER_IMG) * PIX_PER_WG;
    const int y0     = ptbase / W_;

    const int wave   = tid >> 5;
    const int lane   = tid & 31;
    const int mtile  = wave & 1;
    const int pgroup = wave >> 1;
    const bool lo    = lane < 16;
    const int  l15   = lane & 15;
    const int  kbase = lo ? 0 : 16;

    const float* __restrict__ xn = x + (size_t)n * (C_ * HW_);

    // Two pixel tiles per wave; clamp second (last WG of each image is short).
    const int gp0 = ptbase + pgroup * 16 + l15;
    const int gp1 = gp0 + 64;
    const bool v1 = (gp1 < HW_);
    const int g1c = v1 ? gp1 : gp0;

    const int oy0 = gp0 / W_, ox0 = gp0 - oy0 * W_;
    const int oy1 = g1c / W_, ox1 = g1c - oy1 * W_;
    const int hb0 = ((oy0 - y0 + 1) * HALO_W + ox0 + 1) * 32 + kbase;
    const int hb1 = ((oy1 - y0 + 1) * HALO_W + ox1 + 1) * 32 + kbase;

    const int row = mtile * 16 + l15;
    const int awb = row * (KS_*KS_*32) + (lo ? 0 : 16); // A base within slab

    const unsigned w_off0 = lds_byte_offset(&s_w[0][0]);
    const unsigned w_off1 = lds_byte_offset(&s_w[1][0]);

    // Prime the pipeline: slab for cb=0.
    if (wave == 0)
        tdm_load_wslab(w_off0, wpack /* + 0*32 */);

    v8f acc0 = {}, acc1 = {};

    for (int cb = 0; cb < 8; ++cb) {
        const int cbase = cb << 5;

        __syncthreads();          // prev block's LDS reads (halo + old slab) done
        if (wave == 0 && cb < 7)  // DMA next slab while we stage + compute
            tdm_load_wslab((cb & 1) ? w_off0 : w_off1, wpack + (cbase + 32));

        for (int e = tid; e < HALO_ELEMS; e += 256) {
            const int hx = e % HALO_W;          // consecutive tid -> consecutive x
            const int t  = e / HALO_W;
            const int ch = t & 31;
            const int hy = t >> 5;
            const int iy = y0 - 1 + hy;
            const int ix = hx - 1;
            float v = 0.f;
            if ((unsigned)iy < (unsigned)H_ && (unsigned)ix < (unsigned)W_)
                v = xn[(size_t)(cbase + ch) * HW_ + iy * W_ + ix];
            s_t[(hy * HALO_W + hx) * 32 + ch] = f2bf(v);
        }
        if (cb < 7)   // warm L2 for next channel block (global_prefetch_b8)
            __builtin_prefetch(xn + (size_t)(cbase + 32 + (tid & 31)) * HW_ +
                               (y0 ? (y0 - 1) * W_ : 0), 0, 1);

        if (wave == 0) {          // current slab must have landed
            if (cb < 7) { WAIT_TENSORCNT(1); } else { WAIT_TENSORCNT(0); }
        }
        __syncthreads();

        const unsigned short* __restrict__ wbuf = s_w[cb & 1];

        #pragma unroll
        for (int tap = 0; tap < 9; ++tap) {
            const int dy   = tap / 3 - 1;
            const int dx   = tap - (tap / 3) * 3 - 1;
            const int hoff = (dy * HALO_W + dx) * 32;

            // A fragment from LDS slab: line = row*9+tap, 32 bf16 per line.
            ABfrag af;
            const unsigned short* ar = wbuf + awb + tap * 32;
            af.u.lo = *(const uint4*)(ar);
            af.u.hi = *(const uint4*)(ar + 8);

            // B fragments from halo tile (contiguous 32 B per lane half).
            ABfrag b0, b1;
            const unsigned short* p0 = s_t + hb0 + hoff;
            b0.u.lo = *(const uint4*)(p0);
            b0.u.hi = *(const uint4*)(p0 + 8);
            const unsigned short* p1 = s_t + hb1 + hoff;
            b1.u.lo = *(const uint4*)(p1);
            b1.u.hi = *(const uint4*)(p1 + 8);

            acc0 = __builtin_amdgcn_wmma_f32_16x16x32_bf16(
                       false, af.v, false, b0.v, (short)0, acc0, false, false);
            acc1 = __builtin_amdgcn_wmma_f32_16x16x32_bf16(
                       false, af.v, false, b1.v, (short)0, acc1, false, false);
        }
    }

    // Store C: lane l, VGPR r -> M = r + (lane>=16 ? 8 : 0), N = lane%16.
    const int mbase = mtile * 16 + (lo ? 0 : 8);
    {
        float* op = out + ((size_t)n * BUCKET_ + mbase) * HW_ + gp0;
        #pragma unroll
        for (int r = 0; r < 8; ++r)
            op[(size_t)r * HW_] = acc0[r] * (float)(O_ / BUCKET_);
    }
    if (v1) {
        float* op = out + ((size_t)n * BUCKET_ + mbase) * HW_ + gp1;
        #pragma unroll
        for (int r = 0; r < 8; ++r)
            op[(size_t)r * HW_] = acc1[r] * (float)(O_ / BUCKET_);
    }
}

// ---------------------------------------------------------------------------
extern "C" void kernel_launch(void* const* d_in, const int* in_sizes, int n_in,
                              void* d_out, int out_size, void* d_ws, size_t ws_size,
                              hipStream_t stream)
{
    const float* x       = (const float*)d_in[0];
    const float* kernels = (const float*)d_in[1];
    const float* a       = (const float*)d_in[2];
    const int*   table   = (const int*)d_in[3];
    float*       out     = (float*)d_out;

    int* rows_ws = (int*)d_ws;
    unsigned short* wpack = (unsigned short*)((char*)d_ws + 256); // 144 KB bf16

    alsh_select_pack<<<1, 256, 0, stream>>>(x, kernels, a, table, rows_ws, wpack);
    alsh_conv_wmma<<<N_ * WGS_PER_IMG, 256, 0, stream>>>(x, wpack, out);
}
